// S4Layer_26998164423380
// MI455X (gfx1250) — compile-verified, matrix operand order checked by reading, not run on hardware
//
#include <hip/hip_runtime.h>
#include <hip/hip_bf16.h>
#include <math.h>

// ---------------- problem constants ----------------
constexpr int Bb = 4;      // batch
constexpr int Hh = 512;    // d_model
constexpr int Nn = 64;     // d_state/2
constexpr int Ll = 2048;   // seq len
constexpr int L2 = 4096;   // 2*L (circular conv length)

typedef __bf16 bf16_t;
typedef __attribute__((ext_vector_type(16))) __bf16 v16bf;
typedef __attribute__((ext_vector_type(8)))  __bf16 v8bf;
typedef __attribute__((ext_vector_type(8)))  float  v8f;
typedef __attribute__((ext_vector_type(4)))  float  v4f;

typedef __attribute__((ext_vector_type(4))) unsigned int u32x4;
typedef __attribute__((ext_vector_type(8))) int          i32x8;
typedef __attribute__((ext_vector_type(4))) int          i32x4;

// =====================================================================
// Kernel 1: SSM kernel generation.
//   k[c,h,l] = 2*Re( sum_n CB[c,h,n] * exp(dtA[h,n]*l) )
// Stored as reversed circular image R[h][i] = kb[h][(2048 - i) mod 4096]
// so Toeplitz rows become ascending-contiguous windows:
//   A[l][m] = kb[(l-m) mod 4096] = R[2048 + m - l],  index in [1,4095].
// =====================================================================
__global__ void k_gen(const float* __restrict__ log_dt,
                      const float* __restrict__ log_A_real,
                      const float* __restrict__ A_imag,
                      const float* __restrict__ B_re, const float* __restrict__ B_im,
                      const float* __restrict__ C_re, const float* __restrict__ C_im,
                      bf16_t* __restrict__ R) {
    __shared__ float sAr[Nn], sAi[Nn], sC0r[Nn], sC0i[Nn], sC1r[Nn], sC1i[Nn];
    const int h = blockIdx.x;
    const int t = threadIdx.x;
    if (t < Nn) {
        const int n = t;
        float dt  = __expf(log_dt[h]);
        float Are = -__expf(log_A_real[h * Nn + n]);
        float Aim = A_imag[h * Nn + n];
        float dAr = Are * dt, dAi = Aim * dt;
        float er = __expf(dAr);
        float sn, cs; __sincosf(dAi, &sn, &cs);
        float Er = er * cs - 1.0f, Ei = er * sn;           // exp(dtA) - 1
        float den = Are * Are + Aim * Aim;
        float inv = 1.0f / den;
        float qr = (Er * Are + Ei * Aim) * inv;            // (exp(dtA)-1)/A
        float qi = (Ei * Are - Er * Aim) * inv;
        float br = B_re[h * Nn + n], bi = B_im[h * Nn + n];
        float dBr = br * qr - bi * qi, dBi = br * qi + bi * qr;
        float c0r = C_re[h * Nn + n],           c0i = C_im[h * Nn + n];
        float c1r = C_re[Hh * Nn + h * Nn + n], c1i = C_im[Hh * Nn + h * Nn + n];
        sAr[n]  = dAr;  sAi[n] = dAi;
        sC0r[n] = c0r * dBr - c0i * dBi;  sC0i[n] = c0r * dBi + c0i * dBr;
        sC1r[n] = c1r * dBr - c1i * dBi;  sC1i[n] = c1r * dBi + c1i * dBr;
    }
    __syncthreads();
    bf16_t* Rh = R + (size_t)h * L2;
    for (int l = t; l < Ll; l += blockDim.x) {
        float fl = (float)l;
        float a0 = 0.0f, a1 = 0.0f;
        #pragma unroll 4
        for (int n = 0; n < Nn; ++n) {
            float e = __expf(sAr[n] * fl);
            float sn, cs; __sincosf(sAi[n] * fl, &sn, &cs);
            a0 += e * (sC0r[n] * cs - sC0i[n] * sn);
            a1 += e * (sC1r[n] * cs - sC1i[n] * sn);
        }
        Rh[2048 - l]              = (bf16_t)(2.0f * a0);   // k0[l] -> kb[l]
        Rh[(l + 2049) & (L2 - 1)] = (bf16_t)(2.0f * a1);   // k1[l] -> kb[4095-l]
    }
}

// =====================================================================
// Kernel 2: W_out f32 -> bf16
// =====================================================================
__global__ void k_wprep(const float* __restrict__ w, bf16_t* __restrict__ wb) {
    int i = blockIdx.x * 256 + threadIdx.x;
    wb[i] = (bf16_t)w[i];
}

// =====================================================================
// Kernel 3: pre-norm LayerNorm over channels (H) for each (b,l).
// =====================================================================
__global__ void k_layernorm(const float* __restrict__ x,
                            const float* __restrict__ ln_w,
                            const float* __restrict__ ln_b,
                            float* __restrict__ zf, bf16_t* __restrict__ zb) {
    const int bl = blockIdx.x;            // 0 .. B*L-1
    const int b = bl / Ll, l = bl % Ll;
    const int t = threadIdx.x;            // 256 threads, 2 channels each
    __shared__ float red[256];
    const float* xb = x + ((size_t)b * Hh) * Ll + l;
    float v0 = xb[(size_t)t * Ll];
    float v1 = xb[(size_t)(t + 256) * Ll];
    red[t] = v0 + v1;
    __syncthreads();
    for (int o = 128; o > 0; o >>= 1) { if (t < o) red[t] += red[t + o]; __syncthreads(); }
    float mu = red[0] * (1.0f / Hh);
    __syncthreads();
    float d0 = v0 - mu, d1 = v1 - mu;
    red[t] = d0 * d0 + d1 * d1;
    __syncthreads();
    for (int o = 128; o > 0; o >>= 1) { if (t < o) red[t] += red[t + o]; __syncthreads(); }
    float rs = rsqrtf(red[0] * (1.0f / Hh) + 1e-5f);
    float z0 = d0 * rs * ln_w[t]       + ln_b[t];
    float z1 = d1 * rs * ln_w[t + 256] + ln_b[t + 256];
    size_t i0 = ((size_t)b * Hh + t)       * Ll + l;
    size_t i1 = ((size_t)b * Hh + t + 256) * Ll + l;
    zf[i0] = z0;  zf[i1] = z1;
    zb[i0] = (bf16_t)z0;  zb[i1] = (bf16_t)z1;
}

// =====================================================================
// Kernel 4: bidirectional conv as circular-Toeplitz GEMM via WMMA bf16.
//
// Per block (one h, 4 l-tiles): the Tensor Data Mover DMAs the z slice
// z_bf16[0..3][h][0..2047] (2D tile: tile_dim0=2048, tile_dim1=4 rows,
// row stride = H*L elements) into LDS once; all 4 waves' B operands then
// come from aligned ds_load_b128. The Toeplitz A operand reads per-lane
// unaligned 16B windows of the (L0-resident, 8KB/h) reversed kernel image.
// =====================================================================
__global__ void __launch_bounds__(128)
k_conv_wmma(const bf16_t* __restrict__ R, const bf16_t* __restrict__ zb,
            float* __restrict__ yc) {
    __shared__ __align__(16) bf16_t zs[Bb * Ll];     // 16 KB

    const int wave = threadIdx.x >> 5;
    const int lane = threadIdx.x & 31;
    const int h    = blockIdx.x >> 5;          // 512 h's
    const int grp  = blockIdx.x & 31;          // 32 groups of 4 l-tiles
    const int l0   = (grp * 4 + wave) * 16;
    const int row  = lane & 15;                // A row (l) / B col (b)
    const int half = lane >> 4;                // k-group select
    const int l_g  = l0 + row;
    const bool act = row < Bb;
    const bf16_t* Rh = R + (size_t)h * L2;

    // ---- TDM: stage z[0..3][h][:] into LDS (wave 0 issues, all waves wait)
    if (wave == 0) {
        // lds_addr: low 32 bits of the shared-aperture generic address are
        // the LDS byte offset (ISA flat-addressing section).
        unsigned lds_off = (unsigned)(unsigned long long)(&zs[0]);
        unsigned long long ga = (unsigned long long)(zb + (size_t)h * Ll);
        u32x4 g0;
        g0[0] = 1u;                                            // count=1 (valid), user D#
        g0[1] = lds_off;                                       // lds_addr
        g0[2] = (unsigned)(ga & 0xFFFFFFFFull);                // global_addr[31:0]
        g0[3] = (unsigned)((ga >> 32) & 0x1FFFFFFull) | (2u << 30);  // addr[56:32] | type=2
        i32x8 g1;
        g1[0] = (int)(1u << 16);                               // data_size=2B; no mask/pad/iter
        g1[1] = (int)((2048u & 0xFFFFu) << 16);                // tensor_dim0[15:0]=2048
        g1[2] = (int)((4u & 0xFFFFu) << 16);                   // dim0[31:16]=0 | tensor_dim1[15:0]=4
        g1[3] = (int)(2048u << 16);                            // dim1[31:16]=0 | tile_dim0=2048
        g1[4] = (int)(4u);                                     // tile_dim1=4 | tile_dim2=0
        g1[5] = (int)((unsigned)Hh * (unsigned)Ll);            // tensor_dim0_stride[31:0] = H*L elems
        g1[6] = 0;                                             // stride[47:32]=0 | dim1_stride lo=0
        g1[7] = 0;
        i32x4 gz = {0, 0, 0, 0};                               // 2D tensor: groups 2/3 unused
#if defined(__clang_major__) && (__clang_major__ >= 23)
        i32x8 gz8 = {0, 0, 0, 0, 0, 0, 0, 0};
        __builtin_amdgcn_tensor_load_to_lds(g0, g1, gz, gz, gz8, 0);
#else
        __builtin_amdgcn_tensor_load_to_lds(g0, g1, gz, gz, 0);
#endif
        __builtin_amdgcn_s_wait_tensorcnt(0);
    }
    __syncthreads();

    __builtin_prefetch(Rh + (lane << 7), 0, 1);    // warm the 8KB Toeplitz image

    const bf16_t* zrow = zs + (act ? row : 0) * Ll;

    v8f acc = {};
    #pragma unroll 4
    for (int m0 = 0; m0 < Ll; m0 += 32) {
        const int i = 2048 + m0 + 8 * half - l_g;      // in [1, 4072]
        v8bf alo, ahi;
        __builtin_memcpy(&alo, Rh + i,      sizeof(alo));   // unaligned 16B (HW-supported)
        __builtin_memcpy(&ahi, Rh + i + 16, sizeof(ahi));
        v16bf a, bm = {};
        #pragma unroll
        for (int j = 0; j < 8; ++j) { a[j] = alo[j]; a[8 + j] = ahi[j]; }
        if (act) {
            v8bf blo = *(const v8bf*)(zrow + m0 + 8 * half);        // ds_load_b128
            v8bf bhi = *(const v8bf*)(zrow + m0 + 8 * half + 16);
            #pragma unroll
            for (int j = 0; j < 8; ++j) { bm[j] = blo[j]; bm[8 + j] = bhi[j]; }
        }
        acc = __builtin_amdgcn_wmma_f32_16x16x32_bf16(
            false, a, false, bm, (short)0, acc, false, false);
    }
    if (act) {
        // D layout: VGPR g -> M = g + 8*half, N = row. 8 consecutive l's.
        float* out = yc + (((size_t)row * Hh + h) * Ll + l0 + 8 * half);
        v4f o0 = { acc[0], acc[1], acc[2], acc[3] };
        v4f o1 = { acc[4], acc[5], acc[6], acc[7] };
        *(v4f*)(out)     = o0;
        *(v4f*)(out + 4) = o1;
    }
}

// =====================================================================
// Kernel 5: u = gelu(yconv + D*z), transposed to [b][l][h] bf16 via LDS
// =====================================================================
__global__ void k_act_tr(const float* __restrict__ yc, const float* __restrict__ zf,
                         const float* __restrict__ D, bf16_t* __restrict__ ut) {
    __shared__ bf16_t tile[32][33];
    int idx = blockIdx.x;
    const int lb = idx & (Ll / 32 - 1); idx >>= 6;   // L/32 = 64
    const int hb = idx & (Hh / 32 - 1); idx >>= 4;   // H/32 = 16
    const int b  = idx;
    const int h0 = hb * 32, l0 = lb * 32;
    const int t = threadIdx.x;                       // 256
    #pragma unroll
    for (int r = 0; r < 4; ++r) {
        int e = t + r * 256;
        int hl = e >> 5, llc = e & 31;
        int hg = h0 + hl;
        size_t ix = ((size_t)b * Hh + hg) * Ll + (l0 + llc);
        float v = yc[ix] + D[hg] * zf[ix];
        float g = 0.5f * v * (1.0f + tanhf(0.7978845608f * (v + 0.044715f * v * v * v)));
        tile[hl][llc] = (bf16_t)g;
    }
    __syncthreads();
    #pragma unroll
    for (int r = 0; r < 4; ++r) {
        int e = t + r * 256;
        int llc = e >> 5, hl = e & 31;
        ut[((size_t)b * Ll + l0 + llc) * Hh + h0 + hl] = tile[hl][llc];
    }
}

// =====================================================================
// Kernel 6: out = W_out @ u + b_out + x  (full-utilization bf16 WMMA GEMM)
// =====================================================================
__global__ void __launch_bounds__(128)
k_out_gemm(const bf16_t* __restrict__ Wb, const bf16_t* __restrict__ ut,
           const float* __restrict__ b_out, const float* __restrict__ x,
           float* __restrict__ out) {
    const int wave = threadIdx.x >> 5;
    const int lane = threadIdx.x & 31;
    const int tile = blockIdx.x * 4 + wave;   // 16384 tiles
    const int ot = tile & 31;                 // 32 o-tiles
    const int ct = tile >> 5;                 // 512 column tiles
    const int o0 = ot * 16;
    const int col0 = ct * 16;                 // global col = b*L + l
    const int b  = col0 >> 11;
    const int l1 = col0 & (Ll - 1);
    const int row = lane & 15, half = lane >> 4;
    const bf16_t* Arow = Wb + (size_t)(o0 + row) * Hh;
    const bf16_t* Bcol = ut + ((size_t)col0 + row) * Hh;

    v8f acc = {};
    #pragma unroll 4
    for (int k0 = 0; k0 < Hh; k0 += 32) {
        v8bf alo = *(const v8bf*)(Arow + k0 + 8 * half);
        v8bf ahi = *(const v8bf*)(Arow + k0 + 8 * half + 16);
        v8bf blo = *(const v8bf*)(Bcol + k0 + 8 * half);
        v8bf bhi = *(const v8bf*)(Bcol + k0 + 8 * half + 16);
        v16bf a, bm;
        #pragma unroll
        for (int j = 0; j < 8; ++j) {
            a[j]  = alo[j]; a[8 + j]  = ahi[j];
            bm[j] = blo[j]; bm[8 + j] = bhi[j];
        }
        acc = __builtin_amdgcn_wmma_f32_16x16x32_bf16(
            false, a, false, bm, (short)0, acc, false, false);
    }
    const int l = l1 + row;
    #pragma unroll
    for (int g = 0; g < 8; ++g) {
        int o = o0 + g + 8 * half;
        size_t ix = ((size_t)b * Hh + o) * Ll + l;
        out[ix] = acc[g] + b_out[o] + x[ix];
    }
}

// =====================================================================
// Launch
// =====================================================================
extern "C" void kernel_launch(void* const* d_in, const int* in_sizes, int n_in,
                              void* d_out, int out_size, void* d_ws, size_t ws_size,
                              hipStream_t stream) {
    const float* x          = (const float*)d_in[0];
    const float* ln_w       = (const float*)d_in[1];
    const float* ln_b       = (const float*)d_in[2];
    const float* log_dt     = (const float*)d_in[3];
    const float* log_A_real = (const float*)d_in[4];
    const float* A_imag     = (const float*)d_in[5];
    const float* B_re       = (const float*)d_in[6];
    const float* B_im       = (const float*)d_in[7];
    const float* C_re       = (const float*)d_in[8];
    const float* C_im       = (const float*)d_in[9];
    const float* D          = (const float*)d_in[10];
    const float* W_out      = (const float*)d_in[11];
    const float* b_out      = (const float*)d_in[12];
    float* out = (float*)d_out;

    char* ws = (char*)d_ws;
    size_t off = 0;
    bf16_t* R  = (bf16_t*)(ws + off); off += (size_t)Hh * L2 * sizeof(bf16_t);        // 4 MB
    bf16_t* Wb = (bf16_t*)(ws + off); off += (size_t)Hh * Hh * sizeof(bf16_t);        // 0.5 MB
    float*  zf = (float*) (ws + off); off += (size_t)Bb * Hh * Ll * sizeof(float);    // 16.8 MB
    bf16_t* zbuf = (bf16_t*)(ws + off); off += (size_t)Bb * Hh * Ll * sizeof(bf16_t); // 8.4 MB
    float*  yc = (float*) (ws + off); off += (size_t)Bb * Hh * Ll * sizeof(float);    // 16.8 MB
    bf16_t* ut = (bf16_t*)(ws + off); off += (size_t)Bb * Hh * Ll * sizeof(bf16_t);   // 8.4 MB

    k_gen<<<Hh, 256, 0, stream>>>(log_dt, log_A_real, A_imag, B_re, B_im, C_re, C_im, R);
    k_wprep<<<(Hh * Hh) / 256, 256, 0, stream>>>(W_out, Wb);
    k_layernorm<<<Bb * Ll, 256, 0, stream>>>(x, ln_w, ln_b, zf, zbuf);
    k_conv_wmma<<<Hh * 32, 128, 0, stream>>>(R, zbuf, yc);
    k_act_tr<<<Bb * (Hh / 32) * (Ll / 32), 256, 0, stream>>>(yc, zf, D, ut);
    k_out_gemm<<<(Hh / 16) * (Bb * Ll / 16) / 4, 128, 0, stream>>>(Wb, ut, b_out, x, out);
}
